// LSTM_51359218925867
// MI455X (gfx1250) — matrix operand order repaired and checked by reading, not compiled, not run on hardware
//
#include <hip/hip_runtime.h>

typedef __bf16 bf16;
typedef __attribute__((ext_vector_type(16))) __bf16 v16bf;
typedef __attribute__((ext_vector_type(8)))  __bf16 v8bf;
typedef __attribute__((ext_vector_type(8)))  float  v8f;

#define T_SEQ 512
#define NB    64          // batch
#define IND   512         // input dim (layer 0)
#define H     1024        // hidden
#define NL    4           // layers
#define NH    4096        // 4*H (gates)
#define MROWS (T_SEQ*NB)  // 32768

#define KSTEP 64          // K elements staged per pipeline stage
#define LDST  72          // padded LDS row stride (bf16): 144B -> conflict-free, 16B aligned

// ---------------- CDNA5 async global->LDS (ASYNCcnt-tracked, no VGPR roundtrip) ---------
// LDS destination address is taken from a VGPR; LDS offset = low 32 bits of generic __shared__ pointer.
__device__ __forceinline__ void async_ld16(unsigned lds_off, const void* gaddr) {
  asm volatile("global_load_async_to_lds_b128 %0, %1, off"
               :: "v"(lds_off), "v"((unsigned long long)(uintptr_t)gaddr)
               : "memory");
}

// ---------------- WMMA fragment loaders (CDNA5 16-bit layouts) ----------------
// A (16x32 bf16), lds row-major [m][k], row stride LDST:
//  lanes 0-15: M=lane, K groups {0..7,16..23}; lanes 16-31: M=lane-16, K {8..15,24..31}
__device__ __forceinline__ v16bf frag_a(const bf16* lds, int m0, int lane) {
  int r  = m0 + (lane & 15);
  int kb = (lane >> 4) << 3;               // 0 or 8
  const bf16* p = lds + r * LDST + kb;
  v8bf lo = *(const v8bf*)p;               // K = kb .. kb+7
  v8bf hi = *(const v8bf*)(p + 16);        // K = 16+kb .. 16+kb+7
  v16bf a;
#pragma unroll
  for (int i = 0; i < 8; ++i) { a[i] = lo[i]; a[i + 8] = hi[i]; }
  return a;
}

// B (32x16 bf16) staged as [n][k] row-major, stride LDST:
//  lanes 0-15: N=lane, K groups {0..7,16..23}; lanes 16-31: N=lane-16, K {8..15,24..31}
__device__ __forceinline__ v16bf frag_b(const bf16* lds, int n0, int lane) {
  int n  = n0 + (lane & 15);
  int kb = (lane >> 4) << 3;               // 0 or 8
  const bf16* p = lds + n * LDST + kb;
  v8bf lo = *(const v8bf*)p;
  v8bf hi = *(const v8bf*)(p + 16);
  v16bf b;
#pragma unroll
  for (int i = 0; i < 8; ++i) { b[i] = lo[i]; b[i + 8] = hi[i]; }
  return b;
}

// ---------------- conversions ----------------
__global__ void cvt_f32_bf16(const float* __restrict__ in, bf16* __restrict__ out, int n) {
  int i = blockIdx.x * 256 + threadIdx.x;
  if (i < n) out[i] = (bf16)in[i];
}

__global__ void bias_comb(const float* __restrict__ bih, const float* __restrict__ bhh,
                          float* __restrict__ out, int n) {
  int i = blockIdx.x * 256 + threadIdx.x;
  if (i < n) out[i] = bih[i] + bhh[i];
}

// ---------------- input GEMM: gx[M,NH] = A[M,K] * W[NH,K]^T + bias ----------------
// block tile 64x64, 8 waves: (msub 0..3) x (nsub 0..1); 2 accumulators/wave.
// Double-buffered LDS (KSTEP=64) fed by async global->LDS; 4 async instrs/wave/stage.
__global__ void lstm_in_gemm(const bf16* __restrict__ A, const bf16* __restrict__ W,
                             const float* __restrict__ bias, bf16* __restrict__ gx,
                             int K) {
  __shared__ bf16 sA[2][64 * LDST];
  __shared__ bf16 sB[2][64 * LDST];
  const int tid  = threadIdx.x;
  const int lane = tid & 31;
  const int wave = tid >> 5;
  const int m0blk = blockIdx.y * 64;
  const int n0blk = blockIdx.x * 64;
  const int msub = wave >> 1;
  const int nsub = wave & 1;
  v8f acc0 = {}; v8f acc1 = {};

  auto stage = [&](int buf, int k0) {            // 64 rows x 64 k, A and B
#pragma unroll
    for (int i = 0; i < 2; ++i) {
      int cid = tid + i * 256;                   // 0..511
      int row = cid >> 3;                        // 0..63
      int ch  = (cid & 7) << 3;                  // 0..56
      async_ld16((unsigned)(uintptr_t)&sA[buf][row * LDST + ch],
                 A + (size_t)(m0blk + row) * K + k0 + ch);
      async_ld16((unsigned)(uintptr_t)&sB[buf][row * LDST + ch],
                 W + (size_t)(n0blk + row) * K + k0 + ch);
    }
  };

  stage(0, 0);
  int cur = 0;
  for (int k0 = 0; k0 < K; k0 += KSTEP) {
    int knext = (k0 + KSTEP < K) ? (k0 + KSTEP) : 0;   // last stage is dead but harmless
    stage(cur ^ 1, knext);
    asm volatile("s_wait_asynccnt 0x4" ::: "memory");  // current stage complete
    __syncthreads();
#pragma unroll
    for (int kk = 0; kk < KSTEP; kk += 32) {
      v16bf a  = frag_a(&sA[cur][kk], msub * 16, lane);
      v16bf b0 = frag_b(&sB[cur][kk], nsub * 32, lane);
      v16bf b1 = frag_b(&sB[cur][kk], nsub * 32 + 16, lane);
      acc0 = __builtin_amdgcn_wmma_f32_16x16x32_bf16(false, a, false, b0, (short)0, acc0, false, false);
      acc1 = __builtin_amdgcn_wmma_f32_16x16x32_bf16(false, a, false, b1, (short)0, acc1, false, false);
    }
    __syncthreads();
    cur ^= 1;
  }
  const int col0  = n0blk + nsub * 32 + (lane & 15);
  const int rbase = m0blk + msub * 16 + ((lane >> 4) << 3);
#pragma unroll
  for (int r = 0; r < 8; ++r) {
    int row = rbase + r;
    gx[(size_t)row * NH + col0]      = (bf16)(acc0[r] + bias[col0]);
    gx[(size_t)row * NH + col0 + 16] = (bf16)(acc1[r] + bias[col0 + 16]);
  }
}

// ---------------- recurrent step: g = gx[t] + h_prev @ Whh^T; gates; c,h ----------------
// grid: H/32 blocks; block owns hidden cols [jcol, jcol+32) across all 4 gates.
// 8 waves: (msub 0..3 over batch) x (nsub 0..1 over 16-col halves); 4 accumulators/wave.
// Double-buffered LDS (KSTEP=64); 6 async instrs/wave/stage; 16 K-iterations per step.
__global__ void lstm_step(const bf16* __restrict__ gx_t,   // [NB, NH] slice at t
                          const bf16* __restrict__ Whh,    // [NH, H]
                          const bf16* __restrict__ hprev,  // [NB, H] or unused if t==0
                          bf16* __restrict__ hout,         // [NB, H]
                          float* __restrict__ c,           // [NB, H]
                          int t) {
  __shared__ bf16 sA[2][64 * LDST];
  __shared__ bf16 sB[2][128 * LDST];   // 4 gates x 32 rows
  const int tid  = threadIdx.x;
  const int lane = tid & 31;
  const int wave = tid >> 5;
  const int msub = wave >> 1;
  const int nsub = wave & 1;
  const int jcol = blockIdx.x * 32;

  v8f acc[4] = {};

  if (t > 0) {
    auto stage = [&](int buf, int k0) {
#pragma unroll
      for (int i = 0; i < 2; ++i) {              // A: 64 rows x 64 k
        int cid = tid + i * 256;
        int row = cid >> 3;
        int ch  = (cid & 7) << 3;
        async_ld16((unsigned)(uintptr_t)&sA[buf][row * LDST + ch],
                   hprev + (size_t)row * H + k0 + ch);
      }
#pragma unroll
      for (int i = 0; i < 4; ++i) {              // B: 128 rows (4 gates x 32) x 64 k
        int cid  = tid + i * 256;                // 0..1023
        int brow = cid >> 3;                     // 0..127
        int bch  = (cid & 7) << 3;
        int gate = brow >> 5, nl = brow & 31;
        async_ld16((unsigned)(uintptr_t)&sB[buf][brow * LDST + bch],
                   Whh + (size_t)(gate * H + jcol + nl) * H + k0 + bch);
      }
    };

    stage(0, 0);
    int cur = 0;
    for (int k0 = 0; k0 < H; k0 += KSTEP) {
      int knext = (k0 + KSTEP < H) ? (k0 + KSTEP) : 0;
      stage(cur ^ 1, knext);
      asm volatile("s_wait_asynccnt 0x6" ::: "memory");
      __syncthreads();
#pragma unroll
      for (int kk = 0; kk < KSTEP; kk += 32) {
        v16bf a = frag_a(&sA[cur][kk], msub * 16, lane);
#pragma unroll
        for (int g = 0; g < 4; ++g) {
          v16bf b = frag_b(&sB[cur][g * 32 * LDST + kk], nsub * 16, lane);
          acc[g] = __builtin_amdgcn_wmma_f32_16x16x32_bf16(false, a, false, b, (short)0, acc[g], false, false);
        }
      }
      __syncthreads();
      cur ^= 1;
    }
  }

  const int col   = jcol + nsub * 16 + (lane & 15);   // hidden index
  const int rbase = msub * 16 + ((lane >> 4) << 3);   // batch row base
#pragma unroll
  for (int r = 0; r < 8; ++r) {
    int row = rbase + r;
    float gi = acc[0][r] + (float)gx_t[(size_t)row * NH + col];
    float gf = acc[1][r] + (float)gx_t[(size_t)row * NH + H + col];
    float gg = acc[2][r] + (float)gx_t[(size_t)row * NH + 2 * H + col];
    float go = acc[3][r] + (float)gx_t[(size_t)row * NH + 3 * H + col];
    float iv = 1.f / (1.f + __expf(-gi));
    float fv = 1.f / (1.f + __expf(-gf));
    float gv = tanhf(gg);
    float ov = 1.f / (1.f + __expf(-go));
    float cold = (t > 0) ? c[(size_t)row * H + col] : 0.f;
    float cn = fv * cold + iv * gv;
    c[(size_t)row * H + col] = cn;
    hout[(size_t)row * H + col] = (bf16)(ov * tanhf(cn));
  }
}

// ---------------- final FC: out[t] = fc_b + sigmoid(h[t, B-1, :]) . fc_w ----------------
__global__ void lstm_fc(const bf16* __restrict__ hlast, const float* __restrict__ fcw,
                        const float* __restrict__ fcb, float* __restrict__ out) {
  __shared__ float red[256];
  int t = blockIdx.x;
  const bf16* hrow = hlast + ((size_t)t * NB + (NB - 1)) * H;
  float s = 0.f;
  for (int j = threadIdx.x; j < H; j += 256) {
    float hv = (float)hrow[j];
    s += (1.f / (1.f + __expf(-hv))) * fcw[j];
  }
  red[threadIdx.x] = s;
  __syncthreads();
  for (int off = 128; off > 0; off >>= 1) {
    if (threadIdx.x < off) red[threadIdx.x] += red[threadIdx.x + off];
    __syncthreads();
  }
  if (threadIdx.x == 0) out[t] = red[0] + fcb[0];
}

// ---------------- host ----------------
extern "C" void kernel_launch(void* const* d_in, const int* in_sizes, int n_in,
                              void* d_out, int out_size, void* d_ws, size_t ws_size,
                              hipStream_t stream) {
  (void)in_sizes; (void)n_in; (void)out_size; (void)ws_size;
  const float* x    = (const float*)d_in[0];
  const float* wih0 = (const float*)d_in[1];
  const float* wihr = (const float*)d_in[2];
  const float* whh  = (const float*)d_in[3];
  const float* bih  = (const float*)d_in[4];
  const float* bhh  = (const float*)d_in[5];
  const float* fcw  = (const float*)d_in[6];
  const float* fcb  = (const float*)d_in[7];
  float* out = (float*)d_out;

  char* ws = (char*)d_ws;
  size_t off = 0;
  auto alloc = [&](size_t bytes) {
    void* p = ws + off;
    off = (off + bytes + 255) & ~(size_t)255;
    return p;
  };
  bf16* xb    = (bf16*)alloc((size_t)T_SEQ * NB * IND * 2);
  bf16* hb0   = (bf16*)alloc((size_t)T_SEQ * NB * H * 2);
  bf16* hb1   = (bf16*)alloc((size_t)T_SEQ * NB * H * 2);
  bf16* gx    = (bf16*)alloc((size_t)MROWS * NH * 2);
  bf16* wihb  = (bf16*)alloc(((size_t)NH * IND + 3ull * NH * H) * 2);
  bf16* whhb  = (bf16*)alloc((size_t)NL * NH * H * 2);
  float* biasc = (float*)alloc((size_t)NL * NH * 4);
  float* cst   = (float*)alloc((size_t)NB * H * 4);

  auto cvt = [&](const float* src, bf16* dst, size_t n) {
    cvt_f32_bf16<<<(unsigned)((n + 255) / 256), 256, 0, stream>>>(src, dst, (int)n);
  };
  cvt(x, xb, (size_t)T_SEQ * NB * IND);
  cvt(wih0, wihb, (size_t)NH * IND);
  cvt(wihr, wihb + (size_t)NH * IND, 3ull * NH * H);
  cvt(whh, whhb, (size_t)NL * NH * H);
  bias_comb<<<(NL * NH + 255) / 256, 256, 0, stream>>>(bih, bhh, biasc, NL * NH);

  const bf16* layer_in = xb;
  bf16* layer_out = hb0;
  for (int l = 0; l < NL; ++l) {
    int K = (l == 0) ? IND : H;
    const bf16* Wih = (l == 0) ? wihb : (wihb + (size_t)NH * IND + (size_t)(l - 1) * NH * H);
    dim3 ggrid(NH / 64, MROWS / 64);
    lstm_in_gemm<<<ggrid, 256, 0, stream>>>(layer_in, Wih, biasc + (size_t)l * NH, gx, K);

    const bf16* Whh_l = whhb + (size_t)l * NH * H;
    for (int t = 0; t < T_SEQ; ++t) {
      const bf16* hprev = (t == 0) ? (const bf16*)nullptr
                                   : (const bf16*)(layer_out + (size_t)(t - 1) * NB * H);
      lstm_step<<<H / 32, 256, 0, stream>>>(gx + (size_t)t * NB * NH, Whh_l, hprev,
                                            layer_out + (size_t)t * NB * H, cst, t);
    }
    layer_in = layer_out;
    layer_out = (layer_out == hb0) ? hb1 : hb0;
  }
  lstm_fc<<<T_SEQ, 256, 0, stream>>>((const bf16*)layer_in, fcw, fcb, out);
}
